// PerceiverAttention_19559281066087
// MI455X (gfx1250) — matrix operand order repaired and checked by reading, not compiled, not run on hardware
//
#include <hip/hip_runtime.h>

typedef unsigned short u16;
typedef unsigned int u32;
typedef __attribute__((ext_vector_type(16))) __bf16 v16bf;
typedef __attribute__((ext_vector_type(8)))  float  v8f;

union FragBF { uint4 u[2]; v16bf v; };

// --------------------------------------------------------------------------
// Optional CDNA5 async global->LDS path (guarded; falls back to sync staging)
// Builtin proto (from hipcc diagnostic): void(v4i AS1*, v4i AS3*, Imm, Imm)
// --------------------------------------------------------------------------
#if defined(__has_builtin)
#if __has_builtin(__builtin_amdgcn_global_load_async_to_lds_b128) && \
    __has_builtin(__builtin_amdgcn_s_wait_asynccnt)
#define USE_ASYNC_LDS 1
#endif
#endif
#ifndef USE_ASYNC_LDS
#define USE_ASYNC_LDS 0
#endif

#if USE_ASYNC_LDS
typedef int v4i_ __attribute__((vector_size(4 * sizeof(int))));
typedef __attribute__((address_space(1))) v4i_ gv4i;
typedef __attribute__((address_space(3))) v4i_ lv4i;
__device__ __forceinline__ void async_copy16(const void* g, void* l) {
  __builtin_amdgcn_global_load_async_to_lds_b128((gv4i*)g, (lv4i*)l, 0, 0);
}
__device__ __forceinline__ void async_wait0() {
  __builtin_amdgcn_s_wait_asynccnt(0);
}
#endif

__device__ __forceinline__ u16 f2bf(float f) {
  union { float f; u32 u; } c; c.f = f;
  u32 r = c.u + 0x7FFFu + ((c.u >> 16) & 1u);
  return (u16)(r >> 16);
}

// ---------------------------------------------------------------------------
// fp32 -> bf16 elementwise (weights), 4 elems/thread
// ---------------------------------------------------------------------------
__global__ __launch_bounds__(256) void cvt_bf16_kernel(const float* __restrict__ src,
                                                       u16* __restrict__ dst, int n4) {
  int i = (blockIdx.x * 256 + threadIdx.x);
  if (i < n4) {
    float4 v = ((const float4*)src)[i];
    union { u16 h[4]; uint2 u; } o;
    o.h[0] = f2bf(v.x); o.h[1] = f2bf(v.y); o.h[2] = f2bf(v.z); o.h[3] = f2bf(v.w);
    ((uint2*)dst)[i] = o.u;
  }
}

// ---------------------------------------------------------------------------
// LayerNorm over rows of 1024 fp32 -> bf16 (optionally duplicated to 2nd dst)
// ---------------------------------------------------------------------------
__global__ __launch_bounds__(256) void ln_kernel(const float* __restrict__ x,
                                                 const float* __restrict__ g,
                                                 const float* __restrict__ bb,
                                                 u16* __restrict__ dst1,
                                                 u16* __restrict__ dst2,
                                                 int rowsPerBatch, int dstBatchStride,
                                                 int dstRowOff) {
  const int gr = blockIdx.x;
  const int b  = gr / rowsPerBatch;
  const int r  = gr % rowsPerBatch;
  const int t  = threadIdx.x;
  const int lane = t & 31, w = t >> 5;

  float4 v = ((const float4*)(x + (size_t)gr * 1024))[t];
  float s  = v.x + v.y + v.z + v.w;
  float sq = v.x * v.x + v.y * v.y + v.z * v.z + v.w * v.w;
#pragma unroll
  for (int m = 1; m < 32; m <<= 1) { s += __shfl_xor(s, m); sq += __shfl_xor(sq, m); }

  __shared__ float red[2][8];
  if (lane == 0) { red[0][w] = s; red[1][w] = sq; }
  __syncthreads();
  float ts = 0.f, tsq = 0.f;
#pragma unroll
  for (int i = 0; i < 8; ++i) { ts += red[0][i]; tsq += red[1][i]; }
  const float mean = ts * (1.0f / 1024.0f);
  const float var  = tsq * (1.0f / 1024.0f) - mean * mean;
  const float rstd = rsqrtf(var + 1e-5f);

  float4 gg = ((const float4*)g)[t];
  float4 bv = ((const float4*)bb)[t];
  union { u16 h[4]; uint2 u; } o;
  o.h[0] = f2bf((v.x - mean) * rstd * gg.x + bv.x);
  o.h[1] = f2bf((v.y - mean) * rstd * gg.y + bv.y);
  o.h[2] = f2bf((v.z - mean) * rstd * gg.z + bv.z);
  o.h[3] = f2bf((v.w - mean) * rstd * gg.w + bv.w);

  u16* d1 = dst1 + ((size_t)b * dstBatchStride + dstRowOff + r) * 1024;
  ((uint2*)d1)[t] = o.u;
  if (dst2) ((uint2*)(dst2 + (size_t)gr * 1024))[t] = o.u;
}

// ---------------------------------------------------------------------------
// bf16 WMMA GEMM: C[M,N] = A[M,K] * W[K,N]; A,W bf16 row-major.
// Block tile 128x128, 8 waves (4M x 2N), 32x64 per wave, K step 32.
// ---------------------------------------------------------------------------
template <int STORE_F32>
__global__ __launch_bounds__(256) void gemm_wmma_kernel(const u16* __restrict__ A,
                                                        const u16* __restrict__ W,
                                                        void* __restrict__ Cout,
                                                        int M, int N, int K, float scale) {
  const int lane = threadIdx.x & 31;
  const int w    = threadIdx.x >> 5;
  const int wm   = w >> 1;   // 0..3
  const int wn   = w & 1;    // 0..1
  const int bm   = blockIdx.y * 128;
  const int bn   = blockIdx.x * 128;

  __shared__ u16 sA[128 * 32];   // [row][k]
  __shared__ u16 sBT[128 * 32];  // [n][k]  (transposed stage)

  v8f acc[2][4] = {};

  const int lh = lane >> 4;
  const int ll = lane & 15;

  const int arow = threadIdx.x & 127, ahalf = threadIdx.x >> 7;
  const int bk   = threadIdx.x & 31,  bng   = threadIdx.x >> 5;

  for (int k0 = 0; k0 < K; k0 += 32) {
    // stage A tile: 128 rows x 32 k (async path when available)
#if USE_ASYNC_LDS
    {
      const u16* src = A + (size_t)(bm + arow) * K + k0 + ahalf * 16;
      u16* dst = sA + arow * 32 + ahalf * 16;
      async_copy16(src, dst);
      async_copy16(src + 8, dst + 8);
    }
#else
    {
      const uint4* src = (const uint4*)(A + (size_t)(bm + arow) * K + k0 + ahalf * 16);
      uint4 a0 = src[0], a1 = src[1];
      uint4* dst = (uint4*)(sA + arow * 32 + ahalf * 16);
      dst[0] = a0; dst[1] = a1;
    }
#endif
    // stage W tile transposed: read W[k][n] rows, write sBT[n][k]
    {
      const u16* src = W + (size_t)(k0 + bk) * N + bn + bng * 16;
      u16 tmp[16];
      *(uint4*)(tmp)     = ((const uint4*)src)[0];
      *(uint4*)(tmp + 8) = ((const uint4*)src)[1];
#pragma unroll
      for (int j = 0; j < 16; ++j) sBT[(bng * 16 + j) * 32 + bk] = tmp[j];
    }
    // prefetch next K-step tiles into cache
    if (k0 + 32 < K) {
      __builtin_prefetch(A + (size_t)(bm + arow) * K + k0 + 32 + ahalf * 16, 0, 1);
      __builtin_prefetch(W + (size_t)(k0 + 32 + bk) * N + bn + bng * 16, 0, 1);
    }
#if USE_ASYNC_LDS
    async_wait0();
#endif
    __syncthreads();

    FragBF af[2];
#pragma unroll
    for (int mt = 0; mt < 2; ++mt) {
      const u16* p = sA + (wm * 32 + mt * 16 + ll) * 32 + lh * 8;
      af[mt].u[0] = *(const uint4*)p;
      af[mt].u[1] = *(const uint4*)(p + 16);
    }
#pragma unroll
    for (int nt = 0; nt < 4; ++nt) {
      const u16* p = sBT + (wn * 64 + nt * 16 + ll) * 32 + lh * 8;
      FragBF bf;
      bf.u[0] = *(const uint4*)p;
      bf.u[1] = *(const uint4*)(p + 16);
#pragma unroll
      for (int mt = 0; mt < 2; ++mt)
        acc[mt][nt] = __builtin_amdgcn_wmma_f32_16x16x32_bf16(
            false, af[mt].v, false, bf.v, (short)0, acc[mt][nt], false, false);
    }
    __syncthreads();
  }

#pragma unroll
  for (int mt = 0; mt < 2; ++mt)
#pragma unroll
    for (int nt = 0; nt < 4; ++nt)
#pragma unroll
      for (int r = 0; r < 8; ++r) {
        const int m = bm + wm * 32 + mt * 16 + r + 8 * lh;
        const int n = bn + wn * 64 + nt * 16 + ll;
        const float val = acc[mt][nt][r] * scale;
        if (STORE_F32) ((float*)Cout)[(size_t)m * N + n] = val;
        else           ((u16*)Cout)[(size_t)m * N + n]   = f2bf(val);
      }
}

// ---------------------------------------------------------------------------
// Flash attention: q bf16 [B*512,1024], kv bf16 [B*4608,2048] (k|v halves),
// ao bf16 [B*512,1024]. Grid: B*H*4 blocks, 256 thr; wave = 16-query tile.
// ---------------------------------------------------------------------------
__global__ __launch_bounds__(256) void attn_wmma_kernel(const u16* __restrict__ q,
                                                        const u16* __restrict__ kv,
                                                        u16* __restrict__ ao) {
  const int lane = threadIdx.x & 31;
  const int w    = threadIdx.x >> 5;
  const int bx   = blockIdx.x;
  const int qblk = bx & 3;
  const int h    = (bx >> 2) & 15;
  const int b    = bx >> 6;
  const int q0   = qblk * 128 + w * 16;
  const int lh   = lane >> 4, ll = lane & 15;

  __shared__ u16 sK[32 * 64];    // [key][d]
  __shared__ u16 sVT[64 * 32];   // [d][key]
  __shared__ u16 sP[8][16 * 32]; // per-wave P tile [m][key]

  // q fragments (head dim 64, two WMMA K-steps)
  FragBF qf[2];
  {
    const u16* qrow = q + (size_t)(b * 512 + q0 + ll) * 1024 + h * 64;
#pragma unroll
    for (int s = 0; s < 2; ++s) {
      const int base = s * 32 + lh * 8;
      qf[s].u[0] = *(const uint4*)(qrow + base);
      qf[s].u[1] = *(const uint4*)(qrow + base + 16);
    }
  }

  float mrow[8], lrow[8];
#pragma unroll
  for (int r = 0; r < 8; ++r) { mrow[r] = -1e30f; lrow[r] = 0.f; }
  v8f acc[4] = {};

  const size_t kvbase = (size_t)b * 4608 * 2048;
  const int krow = threadIdx.x >> 3;
  const int kdp  = (threadIdx.x & 7) * 8;
  const int vkey = threadIdx.x & 31;
  const int vdg  = (threadIdx.x >> 5) * 8;

  for (int kc = 0; kc < 4608; kc += 32) {
    // stage K chunk [32][64] (async path when available)
#if USE_ASYNC_LDS
    async_copy16(kv + kvbase + (size_t)(kc + krow) * 2048 + h * 64 + kdp,
                 sK + krow * 64 + kdp);
#else
    {
      uint4 d = *(const uint4*)(kv + kvbase + (size_t)(kc + krow) * 2048 + h * 64 + kdp);
      *(uint4*)(sK + krow * 64 + kdp) = d;
    }
#endif
    // stage V chunk transposed -> sVT[d][key]
    {
      u16 tmp[8];
      *(uint4*)tmp =
          *(const uint4*)(kv + kvbase + (size_t)(kc + vkey) * 2048 + 1024 + h * 64 + vdg);
#pragma unroll
      for (int j = 0; j < 8; ++j) sVT[(vdg + j) * 32 + vkey] = tmp[j];
    }
    // prefetch next chunk's K and V rows
    if (kc + 32 < 4608) {
      __builtin_prefetch(kv + kvbase + (size_t)(kc + 32 + krow) * 2048 + h * 64 + kdp, 0, 1);
      __builtin_prefetch(kv + kvbase + (size_t)(kc + 32 + vkey) * 2048 + 1024 + h * 64 + vdg, 0, 1);
    }
#if USE_ASYNC_LDS
    async_wait0();
#endif
    __syncthreads();

    // scores: S = Q * K^T  (16 x 32 as two 16x16 C tiles)
    v8f st[2];
#pragma unroll
    for (int kt = 0; kt < 2; ++kt) {
      v8f c = {};
#pragma unroll
      for (int s = 0; s < 2; ++s) {
        const u16* p = sK + (kt * 16 + ll) * 64 + s * 32 + lh * 8;
        FragBF bf;
        bf.u[0] = *(const uint4*)p;
        bf.u[1] = *(const uint4*)(p + 16);
        c = __builtin_amdgcn_wmma_f32_16x16x32_bf16(false, qf[s].v, false, bf.v,
                                                    (short)0, c, false, false);
      }
      st[kt] = c;
    }

    // online softmax (rows striped across VGPR idx; N across 16 lanes)
    float alpha[8], mnew[8];
#pragma unroll
    for (int r = 0; r < 8; ++r) {
      float mx = fmaxf(st[0][r], st[1][r]);
      mx = fmaxf(mx, __shfl_xor(mx, 1));
      mx = fmaxf(mx, __shfl_xor(mx, 2));
      mx = fmaxf(mx, __shfl_xor(mx, 4));
      mx = fmaxf(mx, __shfl_xor(mx, 8));
      mnew[r]  = fmaxf(mrow[r], mx);
      alpha[r] = __expf(mrow[r] - mnew[r]);
      mrow[r]  = mnew[r];
    }
#pragma unroll
    for (int kt = 0; kt < 2; ++kt)
#pragma unroll
      for (int r = 0; r < 8; ++r) st[kt][r] = __expf(st[kt][r] - mnew[r]);
#pragma unroll
    for (int r = 0; r < 8; ++r) {
      float sum = st[0][r] + st[1][r];
      sum += __shfl_xor(sum, 1);
      sum += __shfl_xor(sum, 2);
      sum += __shfl_xor(sum, 4);
      sum += __shfl_xor(sum, 8);
      lrow[r] = lrow[r] * alpha[r] + sum;
    }
#pragma unroll
    for (int nt = 0; nt < 4; ++nt)
#pragma unroll
      for (int r = 0; r < 8; ++r) acc[nt][r] *= alpha[r];

    // re-layout P (C layout) -> A fragment via per-wave LDS tile
    u16* pw = sP[w];
#pragma unroll
    for (int kt = 0; kt < 2; ++kt)
#pragma unroll
      for (int r = 0; r < 8; ++r)
        pw[(r + 8 * lh) * 32 + kt * 16 + ll] = f2bf(st[kt][r]);

    FragBF pf;
    {
      const u16* pp = pw + ll * 32 + lh * 8;
      pf.u[0] = *(const uint4*)pp;
      pf.u[1] = *(const uint4*)(pp + 16);
    }
    // O += P * V
#pragma unroll
    for (int nt = 0; nt < 4; ++nt) {
      const u16* vp = sVT + (nt * 16 + ll) * 32 + lh * 8;
      FragBF vf;
      vf.u[0] = *(const uint4*)vp;
      vf.u[1] = *(const uint4*)(vp + 16);
      acc[nt] = __builtin_amdgcn_wmma_f32_16x16x32_bf16(false, pf.v, false, vf.v,
                                                        (short)0, acc[nt], false, false);
    }
    __syncthreads();
  }

  // normalize + store bf16
  float inv[8];
#pragma unroll
  for (int r = 0; r < 8; ++r) inv[r] = 1.0f / lrow[r];
#pragma unroll
  for (int nt = 0; nt < 4; ++nt)
#pragma unroll
    for (int r = 0; r < 8; ++r) {
      const int m = q0 + r + 8 * lh;
      const int n = h * 64 + nt * 16 + ll;
      ao[(size_t)(b * 512 + m) * 1024 + n] = f2bf(acc[nt][r] * inv[r]);
    }
}

// ---------------------------------------------------------------------------
extern "C" void kernel_launch(void* const* d_in, const int* in_sizes, int n_in,
                              void* d_out, int out_size, void* d_ws, size_t ws_size,
                              hipStream_t stream) {
  const float* img  = (const float*)d_in[0];  // [4,4096,1024]
  const float* lat  = (const float*)d_in[1];  // [4,512,1024]
  const float* l1g  = (const float*)d_in[2];
  const float* l1b  = (const float*)d_in[3];
  const float* l2g  = (const float*)d_in[4];
  const float* l2b  = (const float*)d_in[5];
  const float* Wq   = (const float*)d_in[6];  // [1024,1024]
  const float* Wkv  = (const float*)d_in[7];  // [1024,2048]
  const float* Wo   = (const float*)d_in[8];  // [1024,1024]
  float* out = (float*)d_out;                 // [4,512,1024]

  // workspace partition (bf16 ushorts), total = 128 MB
  u16* ws    = (u16*)d_ws;
  u16* kvin  = ws;                               // [4*4608, 1024]
  u16* lq    = kvin + (size_t)18432 * 1024;      // [2048, 1024]
  u16* qb    = lq   + (size_t)2048 * 1024;       // [2048, 1024]
  u16* kvb   = qb   + (size_t)2048 * 1024;       // [18432, 2048]
  u16* aob   = kvb  + (size_t)18432 * 2048;      // [2048, 1024]
  u16* wqb   = aob  + (size_t)2048 * 1024;       // [1024, 1024]
  u16* wkvb  = wqb  + (size_t)1024 * 1024;       // [1024, 2048]
  u16* wob   = wkvb + (size_t)1024 * 2048;       // [1024, 1024]

  // weights -> bf16
  cvt_bf16_kernel<<<1024, 256, 0, stream>>>(Wq,  wqb,  1024 * 1024 / 4);
  cvt_bf16_kernel<<<2048, 256, 0, stream>>>(Wkv, wkvb, 1024 * 2048 / 4);
  cvt_bf16_kernel<<<1024, 256, 0, stream>>>(Wo,  wob,  1024 * 1024 / 4);

  // layernorms -> bf16 (image rows into kv_in[0..4095], latents into [4096..4607] + lq)
  ln_kernel<<<16384, 256, 0, stream>>>(img, l1g, l1b, kvin, nullptr, 4096, 4608, 0);
  ln_kernel<<<2048, 256, 0, stream>>>(lat, l2g, l2b, kvin, lq, 512, 4608, 4096);

  // q = lq * Wq  (fold split-sqrt scale: 1/sqrt(64) = 0.125 onto scores via q)
  gemm_wmma_kernel<0><<<dim3(1024 / 128, 2048 / 128), 256, 0, stream>>>(
      lq, wqb, qb, 2048, 1024, 1024, 0.125f);
  // kv = kv_in * Wkv
  gemm_wmma_kernel<0><<<dim3(2048 / 128, 18432 / 128), 256, 0, stream>>>(
      kvin, wkvb, kvb, 18432, 2048, 1024, 1.0f);

  // attention
  attn_wmma_kernel<<<4 * 16 * 4, 256, 0, stream>>>(qb, kvb, aob);

  // out = attn * Wo (fp32 output)
  gemm_wmma_kernel<1><<<dim3(1024 / 128, 2048 / 128), 256, 0, stream>>>(
      aob, wob, out, 2048, 1024, 1024, 1.0f);
}
